// DilatedAttention_full_14319420964898
// MI455X (gfx1250) — matrix operand order repaired and checked by reading, not compiled, not run on hardware
//
#include <hip/hip_runtime.h>

// Dilated attention (B=4, H=8, L=2048, D=64, fp32) for gfx1250 / MI455X.
//
// Flash attention with V_WMMA_F32_16X16X32_BF16 and a bf16 hi/lo split
// (x = hi + lo; x*y ~= hi*hi + hi*lo + lo*hi, 3 WMMAs) for ~fp32 accuracy
// at bf16 matrix rate.
//
//  - Q/K/V converted ONCE to bf16 hi/lo planes in d_ws (Q pre-scaled by
//    log2(e) -> softmax uses native v_exp_f32 exp2; V transposed [bh][d][l]).
//  - LDS layouts make every WMMA fragment two contiguous 16B vector loads
//    (padded row strides: bank-conflict-free, 16B aligned).
//  - 4 waves/block share one (b,h,dr); each wave owns two 16-row Q tiles.
//  - K chunk staging uses CDNA5 GLOBAL_LOAD_ASYNC_TO_LDS_B32 (ASYNCcnt),
//    V staging is phase-split so loads clause with a single wait.

#define NHEADS 8
#define BATCH  4
#define SEQ    2048
#define HD     64
#define BHN    (BATCH * NHEADS)
#define NELEM  ((size_t)BHN * SEQ * HD)
#define LOG2E  1.44269504088896340736f

typedef __bf16 v16bf __attribute__((ext_vector_type(16)));
typedef __bf16 v8bf  __attribute__((ext_vector_type(8)));
typedef float  v8f   __attribute__((ext_vector_type(8)));

static __device__ __forceinline__ __bf16 bf16_rn(float f) {
  unsigned u = __builtin_bit_cast(unsigned, f);
  u += 0x7FFFu + ((u >> 16) & 1u);           // round-to-nearest-even
  unsigned short h = (unsigned short)(u >> 16);
  return __builtin_bit_cast(__bf16, h);
}
static __device__ __forceinline__ float bf16_f32(__bf16 b) {
  unsigned short h = __builtin_bit_cast(unsigned short, b);
  unsigned u = ((unsigned)h) << 16;
  return __builtin_bit_cast(float, u);
}

static __device__ __forceinline__ v8f wmma_bf16(v16bf a, v16bf b, v8f c) {
  return __builtin_amdgcn_wmma_f32_16x16x32_bf16(false, a, false, b,
                                                 (short)0, c, false, false);
}

// Async global->LDS 4B copy per lane (CDNA5). LDS dest address comes from a
// VGPR (byte offset within LDS), global address from a VGPR pair. Tracked by
// ASYNCcnt; pair with s_wait_asynccnt before the block barrier.
static __device__ __forceinline__ void async_g2lds_b32(void* ldsDst,
                                                       const void* gSrc) {
  unsigned ldsOff = (unsigned)(size_t)ldsDst;       // AS(3) ptr -> byte offset
  unsigned long long ga = (unsigned long long)(size_t)gSrc;
  asm volatile("global_load_async_to_lds_b32 %0, %1, off"
               :: "v"(ldsOff), "v"(ga) : "memory");
}
static __device__ __forceinline__ void wait_async0() {
  asm volatile("s_wait_asynccnt 0x0" ::: "memory");
}

// ---- fragment loaders (pure vector loads, zero VALU) -----------------------
static __device__ __forceinline__ v16bf load16(const __bf16* p) {
  v8bf a = *(const v8bf*)p;
  v8bf b = *(const v8bf*)(p + 8);
  return __builtin_shufflevector(a, b, 0, 1, 2, 3, 4, 5, 6, 7,
                                 8, 9, 10, 11, 12, 13, 14, 15);
}
static __device__ __forceinline__ v16bf loadA(const __bf16* row, int h2) {
  v8bf a = *(const v8bf*)(row + 16 * h2);
  v8bf b = *(const v8bf*)(row + 16 * h2 + 8);
  return __builtin_shufflevector(a, b, 0, 1, 2, 3, 4, 5, 6, 7,
                                 8, 9, 10, 11, 12, 13, 14, 15);
}
// f32->bf16 converting fragment builders (fallback path only)
static __device__ __forceinline__ void make_afrag(const float* src, int rowStride,
                                                  int m, int h2, int kBase,
                                                  v16bf& hi, v16bf& lo) {
#pragma unroll
  for (int j = 0; j < 16; ++j) {
    int k = kBase + 16 * h2 + j;
    float f = src[m * rowStride + k];
    __bf16 hb = bf16_rn(f);
    hi[j] = hb;
    lo[j] = bf16_rn(f - bf16_f32(hb));
  }
}
static __device__ __forceinline__ void make_bfrag(const float* base, int strideK,
                                                  int h2, v16bf& hi, v16bf& lo) {
#pragma unroll
  for (int j = 0; j < 16; ++j) {
    float f = base[(16 * h2 + j) * strideK];
    __bf16 hb = bf16_rn(f);
    hi[j] = hb;
    lo[j] = bf16_rn(f - bf16_f32(hb));
  }
}

// ---- one-time plane build: Q (scaled by log2e), K row-major, V transposed --
__global__ __launch_bounds__(256)
void build_planes(const float* __restrict__ Q, const float* __restrict__ K,
                  const float* __restrict__ V,
                  __bf16* __restrict__ Qhi, __bf16* __restrict__ Qlo,
                  __bf16* __restrict__ Khi, __bf16* __restrict__ Klo,
                  __bf16* __restrict__ VThi, __bf16* __restrict__ VTlo) {
  size_t i = (size_t)blockIdx.x * 256 + threadIdx.x;
  if (i >= NELEM) return;
  float qf = Q[i] * LOG2E;                  // softmax done in exp2 domain
  __bf16 qh = bf16_rn(qf);
  Qhi[i] = qh;
  Qlo[i] = bf16_rn(qf - bf16_f32(qh));
  float kf = K[i];
  __bf16 kh = bf16_rn(kf);
  Khi[i] = kh;
  Klo[i] = bf16_rn(kf - bf16_f32(kh));
  float vf = V[i];
  size_t d = i & (HD - 1);
  size_t l = (i / HD) % SEQ;
  size_t bh = i / ((size_t)HD * SEQ);
  size_t ti = (bh * HD + d) * SEQ + l;      // transposed [bh][d][l]
  __bf16 vh = bf16_rn(vf);
  VThi[ti] = vh;
  VTlo[ti] = bf16_rn(vf - bf16_f32(vh));
}

// ---- main kernel: 4 waves/block, 2 Q-tiles/wave, shared K/V staging --------
template <int DR>
__global__ __launch_bounds__(128)
void attn_planes(const __bf16* __restrict__ Qhi, const __bf16* __restrict__ Qlo,
                 const __bf16* __restrict__ Khi, const __bf16* __restrict__ Klo,
                 const __bf16* __restrict__ VThi, const __bf16* __restrict__ VTlo,
                 float* __restrict__ out, int accum) {
  constexpr int LK  = SEQ / DR;
  constexpr int BPB = LK / 128;            // blocks per (b,h): 128 Q rows/block
  constexpr int KS = 72;                   // LDS bf16 row strides (padded:
  constexpr int VS = 40;                   //  conflict-free + 16B aligned)
  constexpr int PS = 40;

  const int tid  = threadIdx.x;
  const int lane = tid & 31;
  const int w    = tid >> 5;
  const int n    = lane & 15;
  const int h2   = lane >> 4;

  const int bh  = blockIdx.x / BPB;
  const int blk = blockIdx.x - bh * BPB;
  const int b   = bh >> 3;
  const int h   = bh & 7;
  const int g   = h / (NHEADS / DR);
  const int qtBase = blk * 8 + 2 * w;

  const __bf16* QhiB = Qhi  + (size_t)bh * SEQ * HD;
  const __bf16* QloB = Qlo  + (size_t)bh * SEQ * HD;
  const __bf16* KhiB = Khi  + (size_t)bh * SEQ * HD;
  const __bf16* KloB = Klo  + (size_t)bh * SEQ * HD;
  const __bf16* VhiB = VThi + (size_t)bh * HD * SEQ;
  const __bf16* VloB = VTlo + (size_t)bh * HD * SEQ;

  __shared__ __align__(16) __bf16 ldsKhi[32 * KS], ldsKlo[32 * KS];
  __shared__ __align__(16) __bf16 ldsVhi[64 * VS], ldsVlo[64 * VS];
  __shared__ __align__(16) __bf16 ldsPhi[8 * 16 * PS], ldsPlo[8 * 16 * PS];

  // Q A-fragments straight from global bf16 planes (one-time 16B gathers)
  v16bf qa_hi[2][2], qa_lo[2][2];
#pragma unroll
  for (int q = 0; q < 2; ++q) {
    int qrow = g + DR * (16 * (qtBase + q) + n);
#pragma unroll
    for (int kk = 0; kk < 2; ++kk) {
      qa_hi[q][kk] = loadA(&QhiB[(size_t)qrow * HD + 32 * kk], h2);
      qa_lo[q][kk] = loadA(&QloB[(size_t)qrow * HD + 32 * kk], h2);
    }
  }

  v8f o[2][4];
#pragma unroll
  for (int q = 0; q < 2; ++q)
#pragma unroll
    for (int t = 0; t < 4; ++t) o[q][t] = {};
  float mrow[2][8], lrow[2][8];
#pragma unroll
  for (int q = 0; q < 2; ++q)
#pragma unroll
    for (int r = 0; r < 8; ++r) { mrow[q][r] = -3.0e38f; lrow[q][r] = 0.0f; }

  for (int c = 0; c < LK / 32; ++c) {
    // ---- K planes via async global->LDS (no VGPR round trip, ASYNCcnt)
    // 32 rows x 64 bf16 = 1024 dwords per plane over 128 threads.
#pragma unroll
    for (int it = 0; it < 8; ++it) {
      int idx = it * 128 + tid;
      int row = idx >> 5;
      int c2  = idx & 31;
      int key = g + DR * (32 * c + row);
      async_g2lds_b32(&ldsKhi[row * KS + c2 * 2],
                      KhiB + (size_t)key * HD + c2 * 2);
      async_g2lds_b32(&ldsKlo[row * KS + c2 * 2],
                      KloB + (size_t)key * HD + c2 * 2);
    }
    // ---- V planes (transposed, b16 elements): phase-split register copy so
    // the 16 loads of a phase clause behind a single wait.
#pragma unroll
    for (int ph = 0; ph < 2; ++ph) {
      __bf16 th[8], tl[8];
#pragma unroll
      for (int it = 0; it < 8; ++it) {
        int idx = (ph * 8 + it) * 128 + tid;
        int d  = idx >> 5;
        int kc = idx & 31;
        size_t gi = (size_t)d * SEQ + g + DR * (32 * c + kc);
        th[it] = VhiB[gi];
        tl[it] = VloB[gi];
      }
#pragma unroll
      for (int it = 0; it < 8; ++it) {
        int idx = (ph * 8 + it) * 128 + tid;
        int d  = idx >> 5;
        int kc = idx & 31;
        ldsVhi[d * VS + kc] = th[it];
        ldsVlo[d * VS + kc] = tl[it];
      }
    }
    wait_async0();           // this wave's async K copies landed in LDS
    __syncthreads();         // publish K/V chunk to all waves

    // ---- S = Q*K^T for both Q tiles; each K fragment feeds 6 WMMAs
    v8f s[2][2];
#pragma unroll
    for (int q = 0; q < 2; ++q)
#pragma unroll
      for (int t = 0; t < 2; ++t) s[q][t] = {};
#pragma unroll
    for (int t = 0; t < 2; ++t) {
#pragma unroll
      for (int kk = 0; kk < 2; ++kk) {
        const int off = (16 * t + n) * KS + 32 * kk + 16 * h2;
        v16bf kbh = load16(&ldsKhi[off]);
        v16bf kbl = load16(&ldsKlo[off]);
#pragma unroll
        for (int q = 0; q < 2; ++q) {
          s[q][t] = wmma_bf16(qa_hi[q][kk], kbh, s[q][t]);
          s[q][t] = wmma_bf16(qa_hi[q][kk], kbl, s[q][t]);
          s[q][t] = wmma_bf16(qa_lo[q][kk], kbh, s[q][t]);
        }
      }
    }

    // ---- online softmax in exp2 domain (Q pre-scaled by log2e)
    float alpha[2][8];
#pragma unroll
    for (int q = 0; q < 2; ++q) {
      __bf16* Ph = &ldsPhi[(w * 2 + q) * (16 * PS)];
      __bf16* Pl = &ldsPlo[(w * 2 + q) * (16 * PS)];
#pragma unroll
      for (int r = 0; r < 8; ++r) {
        float mc = fmaxf(s[q][0][r], s[q][1][r]);
#pragma unroll
        for (int off = 1; off < 16; off <<= 1)
          mc = fmaxf(mc, __shfl_xor(mc, off, 32));
        float mnew = fmaxf(mrow[q][r], mc);
        alpha[q][r] = __builtin_amdgcn_exp2f(mrow[q][r] - mnew);
        float p0 = __builtin_amdgcn_exp2f(s[q][0][r] - mnew);
        float p1 = __builtin_amdgcn_exp2f(s[q][1][r] - mnew);
        float ls = p0 + p1;
#pragma unroll
        for (int off = 1; off < 16; off <<= 1)
          ls += __shfl_xor(ls, off, 32);
        lrow[q][r] = lrow[q][r] * alpha[q][r] + ls;
        mrow[q][r] = mnew;
        int row = 8 * h2 + r;
        __bf16 h0 = bf16_rn(p0), h1 = bf16_rn(p1);
        Ph[row * PS + n]      = h0;
        Ph[row * PS + 16 + n] = h1;
        Pl[row * PS + n]      = bf16_rn(p0 - bf16_f32(h0));
        Pl[row * PS + 16 + n] = bf16_rn(p1 - bf16_f32(h1));
      }
    }

    // ---- O = diag(alpha)*O + P*V ; each V fragment feeds 6 WMMAs
    v16bf pa_h[2], pa_l[2];
#pragma unroll
    for (int q = 0; q < 2; ++q) {
      pa_h[q] = loadA(&ldsPhi[(w * 2 + q) * (16 * PS) + n * PS], h2);
      pa_l[q] = loadA(&ldsPlo[(w * 2 + q) * (16 * PS) + n * PS], h2);
#pragma unroll
      for (int t = 0; t < 4; ++t)
#pragma unroll
        for (int r = 0; r < 8; ++r) o[q][t][r] *= alpha[q][r];
    }
#pragma unroll
    for (int t = 0; t < 4; ++t) {
      const int off = (16 * t + n) * VS + 16 * h2;
      v16bf vbh = load16(&ldsVhi[off]);
      v16bf vbl = load16(&ldsVlo[off]);
#pragma unroll
      for (int q = 0; q < 2; ++q) {
        o[q][t] = wmma_bf16(pa_h[q], vbh, o[q][t]);
        o[q][t] = wmma_bf16(pa_h[q], vbl, o[q][t]);
        o[q][t] = wmma_bf16(pa_l[q], vbh, o[q][t]);
      }
    }
    __syncthreads();     // before next chunk's staging overwrites ldsK/ldsV
  }

  // ---- normalize + scatter to dilated rows; dr=1 writes, dr>1 accumulates
#pragma unroll
  for (int q = 0; q < 2; ++q) {
#pragma unroll
    for (int r = 0; r < 8; ++r) {
      int row = 8 * h2 + r;
      int pos = g + DR * (16 * (qtBase + q) + row);
      float inv = 1.0f / lrow[q][r];
      size_t base = (((size_t)b * SEQ + pos) * NHEADS + h) * HD;
#pragma unroll
      for (int t = 0; t < 4; ++t) {
        float val = o[q][t][r] * inv;
        size_t idx = base + 16 * t + n;
        if (accum) val += out[idx];
        out[idx] = val;
      }
    }
  }
}

// ---- fallback (workspace too small): self-converting, f32 LDS staging -----
template <int DR>
__global__ __launch_bounds__(32)
void attn_fallback(const float* __restrict__ Q, const float* __restrict__ K,
                   const float* __restrict__ V, float* __restrict__ out,
                   int accum) {
  constexpr int LK = SEQ / DR;
  constexpr int NT = LK / 16;
  constexpr int QS = 68, KS = 68, PS = 36;

  const int lane = threadIdx.x;
  const int n = lane & 15, h2 = lane >> 4;
  const int bid = blockIdx.x;
  const int bh = bid / NT, qt = bid - bh * NT;
  const int b = bh >> 3, h = bh & 7;
  const int g = h / (NHEADS / DR);

  const float* Qb = Q + (size_t)bh * SEQ * HD;
  const float* Kb = K + (size_t)bh * SEQ * HD;
  const float* Vb = V + (size_t)bh * SEQ * HD;

  __shared__ __align__(16) float ldsQ[16 * QS];
  __shared__ __align__(16) float ldsK[32 * KS];
  __shared__ __align__(16) float ldsV[32 * KS];
  __shared__ __align__(16) float ldsP[16 * PS];

#pragma unroll 4
  for (int r = 0; r < 16; ++r) {
    const float* src = Qb + (size_t)(g + DR * (16 * qt + r)) * HD;
    *(float2*)&ldsQ[r * QS + lane * 2] = *(const float2*)&src[lane * 2];
  }
  __syncthreads();

  v16bf qa_hi[2], qa_lo[2];
#pragma unroll
  for (int kk = 0; kk < 2; ++kk)
    make_afrag(ldsQ, QS, n, h2, 32 * kk, qa_hi[kk], qa_lo[kk]);

  v8f o[4];
#pragma unroll
  for (int t = 0; t < 4; ++t) o[t] = {};
  float mrow[8], lrow[8];
#pragma unroll
  for (int r = 0; r < 8; ++r) { mrow[r] = -3.0e38f; lrow[r] = 0.0f; }

  for (int c = 0; c < LK / 32; ++c) {
#pragma unroll 4
    for (int r = 0; r < 32; ++r) {
      size_t grow = (size_t)(g + DR * (32 * c + r)) * HD;
      *(float2*)&ldsK[r * KS + lane * 2] = *(const float2*)&Kb[grow + lane * 2];
      *(float2*)&ldsV[r * KS + lane * 2] = *(const float2*)&Vb[grow + lane * 2];
    }
    __syncthreads();

    v8f s[2];
#pragma unroll
    for (int t = 0; t < 2; ++t) {
      s[t] = {};
#pragma unroll
      for (int kk = 0; kk < 2; ++kk) {
        v16bf kbh, kbl;
        make_bfrag(&ldsK[(16 * t + n) * KS + 32 * kk], 1, h2, kbh, kbl);
        s[t] = wmma_bf16(qa_hi[kk], kbh, s[t]);
        s[t] = wmma_bf16(qa_hi[kk], kbl, s[t]);
        s[t] = wmma_bf16(qa_lo[kk], kbh, s[t]);
      }
    }

    float alpha[8];
#pragma unroll
    for (int r = 0; r < 8; ++r) {
      float mc = fmaxf(s[0][r], s[1][r]);
#pragma unroll
      for (int off = 1; off < 16; off <<= 1)
        mc = fmaxf(mc, __shfl_xor(mc, off, 32));
      float mnew = fmaxf(mrow[r], mc);
      alpha[r] = __expf(mrow[r] - mnew);
      float p0 = __expf(s[0][r] - mnew);
      float p1 = __expf(s[1][r] - mnew);
      s[0][r] = p0; s[1][r] = p1;
      float ls = p0 + p1;
#pragma unroll
      for (int off = 1; off < 16; off <<= 1)
        ls += __shfl_xor(ls, off, 32);
      lrow[r] = lrow[r] * alpha[r] + ls;
      mrow[r] = mnew;
    }

#pragma unroll
    for (int r = 0; r < 8; ++r) {
      int row = 8 * h2 + r;
      ldsP[row * PS + n]      = s[0][r];
      ldsP[row * PS + 16 + n] = s[1][r];
    }
    __syncthreads();

    v16bf pah, pal;
    make_afrag(ldsP, PS, n, h2, 0, pah, pal);

#pragma unroll
    for (int t = 0; t < 4; ++t) {
#pragma unroll
      for (int r = 0; r < 8; ++r) o[t][r] *= alpha[r];
    }
#pragma unroll
    for (int t = 0; t < 4; ++t) {
      v16bf vbh, vbl;
      make_bfrag(&ldsV[16 * t + n], KS, h2, vbh, vbl);
      o[t] = wmma_bf16(pah, vbh, o[t]);
      o[t] = wmma_bf16(pah, vbl, o[t]);
      o[t] = wmma_bf16(pal, vbh, o[t]);
    }
    __syncthreads();
  }

#pragma unroll
  for (int r = 0; r < 8; ++r) {
    int row = 8 * h2 + r;
    int pos = g + DR * (16 * qt + row);
    float inv = 1.0f / lrow[r];
    size_t base = (((size_t)b * SEQ + pos) * NHEADS + h) * HD;
#pragma unroll
    for (int t = 0; t < 4; ++t) {
      float val = o[t][r] * inv;
      size_t idx = base + 16 * t + n;
      if (accum) val += out[idx];
      out[idx] = val;
    }
  }
}

extern "C" void kernel_launch(void* const* d_in, const int* in_sizes, int n_in,
                              void* d_out, int out_size, void* d_ws, size_t ws_size,
                              hipStream_t stream) {
  (void)in_sizes; (void)n_in; (void)out_size;
  const float* Q = (const float*)d_in[0];
  const float* K = (const float*)d_in[1];
  const float* V = (const float*)d_in[2];
  float* out = (float*)d_out;

  const size_t need = 6 * NELEM * sizeof(__bf16);   // 48 MB: Q/K/V hi+lo planes

  if (d_ws != nullptr && ws_size >= need) {
    __bf16* Qhi  = (__bf16*)d_ws;
    __bf16* Qlo  = Qhi + NELEM;
    __bf16* Khi  = Qlo + NELEM;
    __bf16* Klo  = Khi + NELEM;
    __bf16* VThi = Klo + NELEM;
    __bf16* VTlo = VThi + NELEM;
    build_planes<<<dim3((unsigned)((NELEM + 255) / 256)), dim3(256), 0, stream>>>(
        Q, K, V, Qhi, Qlo, Khi, Klo, VThi, VTlo);
    attn_planes<1><<<dim3(BHN * (SEQ / 1 / 128)), dim3(128), 0, stream>>>(Qhi, Qlo, Khi, Klo, VThi, VTlo, out, 0);
    attn_planes<2><<<dim3(BHN * (SEQ / 2 / 128)), dim3(128), 0, stream>>>(Qhi, Qlo, Khi, Klo, VThi, VTlo, out, 1);
    attn_planes<4><<<dim3(BHN * (SEQ / 4 / 128)), dim3(128), 0, stream>>>(Qhi, Qlo, Khi, Klo, VThi, VTlo, out, 1);
    attn_planes<8><<<dim3(BHN * (SEQ / 8 / 128)), dim3(128), 0, stream>>>(Qhi, Qlo, Khi, Klo, VThi, VTlo, out, 1);
  } else {
    attn_fallback<1><<<dim3(BHN * (SEQ / 1 / 16)), dim3(32), 0, stream>>>(Q, K, V, out, 0);
    attn_fallback<2><<<dim3(BHN * (SEQ / 2 / 16)), dim3(32), 0, stream>>>(Q, K, V, out, 1);
    attn_fallback<4><<<dim3(BHN * (SEQ / 4 / 16)), dim3(32), 0, stream>>>(Q, K, V, out, 1);
    attn_fallback<8><<<dim3(BHN * (SEQ / 8 / 16)), dim3(32), 0, stream>>>(Q, K, V, out, 1);
  }
}
